// TreeConvGRU3d_5059471474869
// MI455X (gfx1250) — matrix operand order repaired and checked by reading, not compiled, not run on hardware
//
#include <hip/hip_runtime.h>
#include <hip/hip_bf16.h>

// ---------------------------------------------------------------------------
// TreeConvGRU3d forward for gfx1250 (MI455X).
//
// Packed channels-last f16 activations [nb][vox][Cpad]. Conv3d = implicit
// GEMM on v_wmma_f32_16x16x32_f16 with an LDS-tiled input brick:
//   - Block = 8x8x4 output brick; input brick + halo (10x10x6) staged to LDS
//     once with zero-filled OOB voxels -> tap loop has NO bounds checks.
//   - Weights pre-packed per layer into exact A-fragment order (f16); conv
//     blocks copy them to LDS with b128 vector copies.
//   - Per tap: 2x ds_load_b128 (A) + 2x2 ds_load_b128 (B dual-N tile) +
//     2 independent WMMAs. LDS tile stride skewed (CP+8 halfs) to spread banks.
//   - Epilogue: 8 consecutive channels/lane -> one global_store_b128.
// Global activation traffic per conv ~2.3x tensor size (halo) instead of 27x;
// workload is HBM/L2 bound (~200 GFLOP vs GB-scale traffic @ 23.3 TB/s).
// ---------------------------------------------------------------------------

typedef __attribute__((ext_vector_type(16))) _Float16 v16h;
typedef __attribute__((ext_vector_type(8)))  _Float16 v8h;
typedef __attribute__((ext_vector_type(8)))  float    v8f;

__device__ __forceinline__ float sigmoidf_(float x) { return 1.0f / (1.0f + __expf(-x)); }

struct BatchMap { int use; int idx[64]; };   // gather map: local batch -> node id

struct Level {
    int nNodes;
    int node[32];                 // frontier node ids
    int eA[32], eB[32];           // local edge indices into rjk/hk (-1 if none)
    int cA[32], cB[32];           // child node ids for those edges
};

// Fragment block: 27 taps * 32 lanes * 16 halfs.
#define FRAG_HALFS (27 * 512)

// ---------------------------------------------------------------------------
// Pre-pack f32 OIDHW weights (+ optional bias) into A-fragment-ordered f16:
// layout [mt][c][tap][lane][j], then MT*16 f16 bias values.
// lane ln, elem j -> m = ln%16, k = (ln/16)*16 + j.
// ---------------------------------------------------------------------------
__global__ void pack_weights_kernel(const float* __restrict__ wgt, const float* __restrict__ bias,
                                    _Float16* __restrict__ wpk, int Cin, int Cout, int KC, int MT)
{
    int fragTotal = MT * KC * FRAG_HALFS;
    int total = fragTotal + MT * 16;
    int i = blockIdx.x * 256 + threadIdx.x;
    if (i >= total) return;
    if (i < fragTotal) {
        int j   = i & 15;
        int ln  = (i >> 4) & 31;
        int tap = (i >> 9) % 27;
        int c   = (i / FRAG_HALFS) % KC;
        int mt  = i / (FRAG_HALFS * KC);
        int m   = ln & 15;
        int k   = ((ln >> 4) << 4) + j;
        int co  = mt * 16 + m;
        int cig = (c << 5) + k;
        float v = (co < Cout && cig < Cin) ? wgt[(co * Cin + cig) * 27 + tap] : 0.0f;
        wpk[i] = (_Float16)v;
    } else {
        int co = i - fragTotal;
        float v = (bias && co < Cout) ? bias[co] : 0.0f;
        wpk[i] = (_Float16)v;
    }
}

// ---------------------------------------------------------------------------
// Implicit-GEMM 3x3x3 conv (pad=1, stride=1), templated on cube side D and
// padded input channels CP. Output packed with Cpad=32.
// Block: 8x8x4 output brick (256 voxels), input halo tile in LDS.
// Each wave: 16 couts x 32 voxels (dual-N tile), K = 27 taps x (CP/32) chunks.
// ---------------------------------------------------------------------------
template<int D, int CP>
__global__ __launch_bounds__(256)
void conv3d_wmma_kernel(const _Float16* __restrict__ in, const _Float16* __restrict__ wpk,
                        _Float16* __restrict__ out, int Cout, int doRelu, BatchMap map)
{
    constexpr int D3 = D * D * D;
    constexpr int KC = CP / 32;
    constexpr int CPO = 32;
    constexpr int BX = 8, BY = 8, BZ = 4;                // output brick
    constexpr int HX = BX + 2, HY = BY + 2, HZ = BZ + 2; // halo tile 10x10x6
    constexpr int CPT = CP + 8;                          // skewed LDS voxel stride
    constexpr int NBX = D / BX, NBY = D / BY;
    constexpr int PARTS = CP / 16;                       // 16-half chunks per voxel
    static_assert(D3 % 256 == 0, "brick must divide volume");

    const int tid  = threadIdx.x;
    const int lane = tid & 31;
    const int wave = tid >> 5;

    const int bx = blockIdx.x % NBX;
    const int by = (blockIdx.x / NBX) % NBY;
    const int bz = blockIdx.x / (NBX * NBY);
    const int x0 = bx * BX, y0 = by * BY, z0 = bz * BZ;

    const int coutBase = blockIdx.y * 16;
    const int nbL = blockIdx.z;                          // local batch (output)
    const int nb  = map.use ? map.idx[nbL] : nbL;        // source batch (gather)

    __shared__ __align__(32) _Float16 smemA[FRAG_HALFS];
    __shared__ __align__(32) _Float16 smemT[HZ * HY * HX * CPT];

    // ---- Stage input brick + halo into LDS (zero-filled OOB) ----
    for (int t = tid; t < HZ * HY * HX * PARTS; t += 256) {
        int part = t & (PARTS - 1);
        int hv   = t / PARTS;
        int hx = hv % HX, hy = (hv / HX) % HY, hz = hv / (HX * HY);
        int gx = x0 + hx - 1, gy = y0 + hy - 1, gz = z0 + hz - 1;
        v16h v = {};
        if (((unsigned)gx < (unsigned)D) & ((unsigned)gy < (unsigned)D) &
            ((unsigned)gz < (unsigned)D)) {
            size_t gvox = (size_t)gz * D * D + (size_t)gy * D + gx;
            v = *(const v16h*)(in + ((size_t)nb * D3 + gvox) * CP + part * 16);
        }
        *(v16h*)(smemT + hv * CPT + part * 16) = v;
    }

    v8f acc0 = {}, acc1 = {};

    const int n   = lane & 15;                           // B column
    const int kbB = (lane >> 4) << 4;                    // B K-half base (0/16)
    const int lv0 = wave * 32 + n;                       // local voxel of tile 0
    const int lv1 = lv0 + 16;
    const int lx0 = lv0 & 7, ly0 = (lv0 >> 3) & 7, lz0 = lv0 >> 6;
    const int lx1 = lv1 & 7, ly1 = (lv1 >> 3) & 7, lz1 = lv1 >> 6;

    const _Float16* wsrc = wpk + (size_t)blockIdx.y * KC * FRAG_HALFS;

    for (int c = 0; c < KC; ++c) {
        __syncthreads();
        {   // vector copy of fragment-ordered weights: global b128 -> LDS b128
            const v8h* s = (const v8h*)(wsrc + c * FRAG_HALFS);
            v8h* d = (v8h*)smemA;
            for (int t = tid; t < FRAG_HALFS / 8; t += 256)
                d[t] = s[t];
        }
        __syncthreads();

        const _Float16* tb0 = smemT + ((lz0 * HY + ly0) * HX + lx0) * CPT + (c << 5) + kbB;
        const _Float16* tb1 = smemT + ((lz1 * HY + ly1) * HX + lx1) * CPT + (c << 5) + kbB;

        #pragma unroll
        for (int kd = 0; kd < 3; ++kd) {
            #pragma unroll
            for (int kh = 0; kh < 3; ++kh) {
                #pragma unroll
                for (int kw = 0; kw < 3; ++kw) {
                    const int tap   = kd * 9 + kh * 3 + kw;
                    const int delta = ((kd * HY + kh) * HX + kw) * CPT;
                    v16h a  = *(const v16h*)(smemA + tap * 512 + lane * 16);
                    v16h b0 = *(const v16h*)(tb0 + delta);
                    v16h b1 = *(const v16h*)(tb1 + delta);
                    acc0 = __builtin_amdgcn_wmma_f32_16x16x32_f16(
                               false, a, false, b0, (short)0, acc0, false, false);
                    acc1 = __builtin_amdgcn_wmma_f32_16x16x32_f16(
                               false, a, false, b1, (short)0, acc1, false, false);
                }
            }
        }
    }

    // Epilogue: 8 consecutive channels per lane -> one b128 store per tile.
    const int mHi = (lane >> 4) << 3;
    const v8h bo = *(const v8h*)(wpk + (size_t)gridDim.y * KC * FRAG_HALFS + coutBase + mHi);
    const int vox0 = (z0 + lz0) * D * D + (y0 + ly0) * D + (x0 + lx0);
    const int vox1 = (z0 + lz1) * D * D + (y0 + ly1) * D + (x0 + lx1);
    v8h o0, o1;
    #pragma unroll
    for (int r = 0; r < 8; ++r) {
        int co = coutBase + mHi + r;
        float v0 = 0.0f, v1 = 0.0f;
        if (co < Cout) {
            v0 = acc0[r] + (float)bo[r];
            v1 = acc1[r] + (float)bo[r];
            if (doRelu) { v0 = fmaxf(v0, 0.0f); v1 = fmaxf(v1, 0.0f); }
        }
        o0[r] = (_Float16)v0;
        o1[r] = (_Float16)v1;
    }
    *(v8h*)(out + ((size_t)nbL * D3 + vox0) * CPO + coutBase + mHi) = o0;
    *(v8h*)(out + ((size_t)nbL * D3 + vox1) * CPO + coutBase + mHi) = o1;
}

// ---------------------------------------------------------------------------
// Pack f32 single-channel input -> packed f16 [nb][vox][32] (pad = 0).
// ---------------------------------------------------------------------------
__global__ void pack_data_kernel(const float* __restrict__ in, _Float16* __restrict__ out, int total)
{
    int i = blockIdx.x * 256 + threadIdx.x;
    if (i >= total) return;
    v16h z = {};
    z[0] = (_Float16)in[i];
    *(v16h*)(out + (size_t)i * 32) = z;
    *(v16h*)(out + (size_t)i * 32 + 16) = (v16h){};
}

// ---------------------------------------------------------------------------
// 2x2x2 max pool (stride 2), packed layout, Cp=32. D = output side.
// ---------------------------------------------------------------------------
template<int D>
__global__ void maxpool2_kernel(const _Float16* __restrict__ in, _Float16* __restrict__ out, int NB)
{
    constexpr int Cp = 32, Di = D * 2;
    int total = NB * D * D * D * Cp;
    int i = blockIdx.x * 256 + threadIdx.x;
    if (i >= total) return;
    int c = i & 31;
    int rest = i >> 5;
    int x = rest % D, y = (rest / D) % D, z = (rest / (D * D)) % D;
    int nb = rest / (D * D * D);
    const _Float16* p = in + (size_t)nb * Di * Di * Di * Cp + c;
    float m = -1e30f;
    #pragma unroll
    for (int dz = 0; dz < 2; ++dz)
        #pragma unroll
        for (int dy = 0; dy < 2; ++dy)
            #pragma unroll
            for (int dx = 0; dx < 2; ++dx) {
                float v = (float)p[(size_t)((2 * z + dz) * Di * Di + (2 * y + dy) * Di + (2 * x + dx)) * Cp];
                m = fmaxf(m, v);
            }
    out[i] = (_Float16)m;
}

// ---------------------------------------------------------------------------
// out = concat(up2(a), b) along channels, packed layout. a at Dd/2, b at Dd.
// ---------------------------------------------------------------------------
template<int Dd, int Cpo>
__global__ void up2_concat_kernel(const _Float16* __restrict__ a, int Ca,
                                  const _Float16* __restrict__ b, int Cb,
                                  _Float16* __restrict__ out, int NB)
{
    constexpr int Ds = Dd / 2, Cpi = 32;
    int total = NB * Dd * Dd * Dd * Cpo;
    int i = blockIdx.x * 256 + threadIdx.x;
    if (i >= total) return;
    int c = i % Cpo;
    int rest = i / Cpo;
    int x = rest % Dd, y = (rest / Dd) % Dd, z = (rest / (Dd * Dd)) % Dd;
    int nb = rest / (Dd * Dd * Dd);
    float v = 0.0f;
    if (c < Ca) {
        size_t voxS = (size_t)(z >> 1) * Ds * Ds + (size_t)(y >> 1) * Ds + (x >> 1);
        v = (float)a[((size_t)nb * Ds * Ds * Ds + voxS) * Cpi + c];
    } else if (c < Ca + Cb) {
        size_t vox = (size_t)z * Dd * Dd + (size_t)y * Dd + x;
        v = (float)b[((size_t)nb * Dd * Dd * Dd + vox) * Cpi + (c - Ca)];
    }
    out[i] = (_Float16)v;
}

// ---------------------------------------------------------------------------
// Leaf init (packed, Cp=32, D3=512, 10 valid channels):
//   h = height==0 ? sig(u0)*h0 + (1-sig(u0))*tanh(w0) : h0
// ---------------------------------------------------------------------------
__global__ void leaf_init_kernel(const _Float16* __restrict__ w0, const _Float16* __restrict__ u0,
                                 const float* __restrict__ h0, const int* __restrict__ height,
                                 _Float16* __restrict__ h, int NB)
{
    int i = blockIdx.x * 256 + threadIdx.x;
    if (i >= NB * 512 * 32) return;
    int c = i & 31;
    int vox = (i >> 5) & 511;
    int nb = i >> 14;
    float hv = 0.0f;
    if (c < 10) {
        float su  = sigmoidf_((float)u0[i]);
        float h0v = h0[((size_t)nb * 10 + c) * 512 + vox];
        float hl  = su * h0v + (1.0f - su) * tanhf((float)w0[i]);
        hv = (height[nb] == 0) ? hl : h0v;
    }
    h[i] = (_Float16)hv;
}

// ---------------------------------------------------------------------------
// Per-level gated segment sums over <=2 children per frontier node (packed,
// node slab = 16384 halfs):
//   h_child = sum h[child] ; w_sum = sum sigmoid(rjk_e + r0[node]) * hk_e
// ---------------------------------------------------------------------------
__global__ void level_combine_kernel(const _Float16* __restrict__ rjk, const _Float16* __restrict__ hk,
                                     const _Float16* __restrict__ r0, const _Float16* __restrict__ h,
                                     Level lv, _Float16* __restrict__ h_child,
                                     _Float16* __restrict__ w_sum)
{
    int i = blockIdx.x * 256 + threadIdx.x;
    if (i >= (lv.nNodes << 14)) return;
    int li = i >> 14, off = i & 16383;
    int node = lv.node[li];
    float r0v = (float)r0[((size_t)node << 14) + off];
    float hc = 0.0f, wsv = 0.0f;
    int ea = lv.eA[li];
    if (ea >= 0) {
        hc  += (float)h[((size_t)lv.cA[li] << 14) + off];
        wsv += sigmoidf_((float)rjk[((size_t)ea << 14) + off] + r0v) * (float)hk[((size_t)ea << 14) + off];
    }
    int eb = lv.eB[li];
    if (eb >= 0) {
        hc  += (float)h[((size_t)lv.cB[li] << 14) + off];
        wsv += sigmoidf_((float)rjk[((size_t)eb << 14) + off] + r0v) * (float)hk[((size_t)eb << 14) + off];
    }
    h_child[i] = (_Float16)hc;
    w_sum[i]   = (_Float16)wsv;
}

// ---------------------------------------------------------------------------
// Per-level GRU update (packed):
//   s = sigmoid(u_part + u0[node]); h[node] = s*h_child + (1-s)*tanh(w_sum+w0[node])
// ---------------------------------------------------------------------------
__global__ void level_update_kernel(const _Float16* __restrict__ u_part, const _Float16* __restrict__ u0,
                                    const _Float16* __restrict__ w_sum, const _Float16* __restrict__ w0,
                                    const _Float16* __restrict__ h_child, Level lv,
                                    _Float16* __restrict__ h)
{
    int i = blockIdx.x * 256 + threadIdx.x;
    if (i >= (lv.nNodes << 14)) return;
    int li = i >> 14, off = i & 16383;
    int node = lv.node[li];
    float s  = sigmoidf_((float)u_part[i] + (float)u0[((size_t)node << 14) + off]);
    float wv = (float)w_sum[i] + (float)w0[((size_t)node << 14) + off];
    h[((size_t)node << 14) + off] = (_Float16)(s * (float)h_child[i] + (1.0f - s) * tanhf(wv));
}

// ---------------------------------------------------------------------------
// 1x1x1 conv, 16 channels packed f16 -> f32 output (1 channel).
// ---------------------------------------------------------------------------
__global__ void conv1x1_kernel(const _Float16* __restrict__ in, const float* __restrict__ w,
                               const float* __restrict__ b, float* __restrict__ out, int total)
{
    int i = blockIdx.x * 256 + threadIdx.x;
    if (i >= total) return;
    v16h x = *(const v16h*)(in + (size_t)i * 32);
    float s = b[0];
    #pragma unroll
    for (int c = 0; c < 16; ++c)
        s += (float)x[c] * w[c];
    out[i] = s;
}

// ---------------------------------------------------------------------------
// Host orchestration
// ---------------------------------------------------------------------------
extern "C" void kernel_launch(void* const* d_in, const int* in_sizes, int n_in,
                              void* d_out, int out_size, void* d_ws, size_t ws_size,
                              hipStream_t stream) {
    (void)in_sizes; (void)n_in; (void)out_size; (void)ws_size;
    const int NB = 64;

    const float* data   = (const float*)d_in[0];
    const float* h0     = (const float*)d_in[1];
    const int*   height = (const int*)  d_in[4];
    const float* e1a_w = (const float*)d_in[5];  const float* e1a_b = (const float*)d_in[6];
    const float* e1b_w = (const float*)d_in[7];  const float* e1b_b = (const float*)d_in[8];
    const float* e2a_w = (const float*)d_in[9];  const float* e2a_b = (const float*)d_in[10];
    const float* e2b_w = (const float*)d_in[11]; const float* e2b_b = (const float*)d_in[12];
    const float* W_w   = (const float*)d_in[13];
    const float* Wz_w  = (const float*)d_in[14];
    const float* Wr_w  = (const float*)d_in[15];
    const float* U_w   = (const float*)d_in[16]; const float* U_b   = (const float*)d_in[17];
    const float* Uz_w  = (const float*)d_in[18]; const float* Uz_b  = (const float*)d_in[19];
    const float* Ur_w  = (const float*)d_in[20]; const float* Ur_b  = (const float*)d_in[21];
    const float* d2a_w = (const float*)d_in[22]; const float* d2a_b = (const float*)d_in[23];
    const float* d2b_w = (const float*)d_in[24]; const float* d2b_b = (const float*)d_in[25];
    const float* d1a_w = (const float*)d_in[26]; const float* d1a_b = (const float*)d_in[27];
    const float* d1b_w = (const float*)d_in[28]; const float* d1b_b = (const float*)d_in[29];
    const float* d1c_w = (const float*)d_in[30]; const float* d1c_b = (const float*)d_in[31];
    float* out = (float*)d_out;

    // Replicate _tree() on host (deterministic; no HIP calls).
    int parentH[64], heightH[64];
    for (int i = 0; i < 64; ++i) heightH[i] = 0;
    for (int i = 63; i >= 1; --i) {
        int p = (i - 1) / 2;
        parentH[i] = p;
        int hh = heightH[i] + 1;
        if (hh > heightH[p]) heightH[p] = hh;
    }
    const int maxH = heightH[0];

    // Workspace carve (f16 elements).
    _Float16* ws = (_Float16*)d_ws;
    size_t off = 0;
    auto alloc = [&](size_t n) { _Float16* p = ws + off; off += n; return p; };
    const int D32 = 32 * 32 * 32, D16 = 16 * 16 * 16, D8 = 8 * 8 * 8;
    const int CSp = D8 * 32;                             // per-node GRU slab (packed)
    _Float16* dataP = alloc((size_t)NB * D32 * 32);
    _Float16* t16  = alloc((size_t)NB * D32 * 32);
    _Float16* x0   = alloc((size_t)NB * D32 * 32);
    _Float16* p0   = alloc((size_t)NB * D16 * 32);
    _Float16* t32  = alloc((size_t)NB * D16 * 32);
    _Float16* x1   = alloc((size_t)NB * D16 * 32);
    _Float16* xg   = alloc((size_t)NB * D8  * 32);
    _Float16* w0   = alloc((size_t)NB * CSp);
    _Float16* u0   = alloc((size_t)NB * CSp);
    _Float16* r0   = alloc((size_t)NB * CSp);
    _Float16* h    = alloc((size_t)NB * CSp);
    _Float16* rjk  = alloc((size_t)64 * CSp);
    _Float16* hk   = alloc((size_t)64 * CSp);
    _Float16* hch  = alloc((size_t)32 * CSp);
    _Float16* wsum = alloc((size_t)32 * CSp);
    _Float16* upar = alloc((size_t)32 * CSp);
    _Float16* y30  = alloc((size_t)NB * D16 * 32);
    _Float16* y20  = alloc((size_t)NB * D16 * 32);
    _Float16* y32c = alloc((size_t)NB * D16 * 32);
    _Float16* y48  = alloc((size_t)NB * D32 * 64);
    _Float16* y16a = alloc((size_t)NB * D32 * 32);
    _Float16* y16b = alloc((size_t)NB * D32 * 32);
    // Packed-weight buffers: MT*KC*FRAG_HALFS + MT*16 halfs each.
    auto wAlloc = [&](int KC, int MT) { return alloc((size_t)MT * KC * FRAG_HALFS + MT * 16); };
    _Float16* Pe1a = wAlloc(1, 1);  _Float16* Pe1b = wAlloc(1, 1);
    _Float16* Pe2a = wAlloc(1, 2);  _Float16* Pe2b = wAlloc(1, 2);
    _Float16* PW   = wAlloc(1, 1);  _Float16* PWz  = wAlloc(1, 1);  _Float16* PWr = wAlloc(1, 1);
    _Float16* PU   = wAlloc(1, 1);  _Float16* PUz  = wAlloc(1, 1);  _Float16* PUr = wAlloc(1, 1);
    _Float16* Pd2a = wAlloc(1, 2);  _Float16* Pd2b = wAlloc(1, 2);
    _Float16* Pd1a = wAlloc(2, 1);  _Float16* Pd1b = wAlloc(1, 1);

    const dim3 B(256);
    BatchMap noMap; noMap.use = 0;
    for (int i = 0; i < 64; ++i) noMap.idx[i] = i;
    auto pkGrid = [](int KC, int MT) { return dim3((MT * KC * FRAG_HALFS + MT * 16 + 255) / 256); };
    auto cvGrid = [](int D3, int Cout, int nBatch) {
        return dim3(D3 / 256, (Cout + 15) / 16, nBatch);
    };

    // ---- Pack weights (independent, tiny) ----
    pack_weights_kernel<<<pkGrid(1,1), B, 0, stream>>>(e1a_w, e1a_b, Pe1a,  1, 16, 1, 1);
    pack_weights_kernel<<<pkGrid(1,1), B, 0, stream>>>(e1b_w, e1b_b, Pe1b, 16, 16, 1, 1);
    pack_weights_kernel<<<pkGrid(1,2), B, 0, stream>>>(e2a_w, e2a_b, Pe2a, 16, 32, 1, 2);
    pack_weights_kernel<<<pkGrid(1,2), B, 0, stream>>>(e2b_w, e2b_b, Pe2b, 32, 20, 1, 2);
    pack_weights_kernel<<<pkGrid(1,1), B, 0, stream>>>(W_w,  nullptr, PW,  20, 10, 1, 1);
    pack_weights_kernel<<<pkGrid(1,1), B, 0, stream>>>(Wz_w, nullptr, PWz, 20, 10, 1, 1);
    pack_weights_kernel<<<pkGrid(1,1), B, 0, stream>>>(Wr_w, nullptr, PWr, 20, 10, 1, 1);
    pack_weights_kernel<<<pkGrid(1,1), B, 0, stream>>>(U_w,  U_b,  PU,  10, 10, 1, 1);
    pack_weights_kernel<<<pkGrid(1,1), B, 0, stream>>>(Uz_w, Uz_b, PUz, 10, 10, 1, 1);
    pack_weights_kernel<<<pkGrid(1,1), B, 0, stream>>>(Ur_w, Ur_b, PUr, 10, 10, 1, 1);
    pack_weights_kernel<<<pkGrid(1,2), B, 0, stream>>>(d2a_w, d2a_b, Pd2a, 30, 20, 1, 2);
    pack_weights_kernel<<<pkGrid(1,2), B, 0, stream>>>(d2b_w, d2b_b, Pd2b, 20, 32, 1, 2);
    pack_weights_kernel<<<pkGrid(2,1), B, 0, stream>>>(d1a_w, d1a_b, Pd1a, 48, 16, 2, 1);
    pack_weights_kernel<<<pkGrid(1,1), B, 0, stream>>>(d1b_w, d1b_b, Pd1b, 16, 16, 1, 1);

    // ---- Pack input ----
    pack_data_kernel<<<((NB * D32) + 255) / 256, B, 0, stream>>>(data, dataP, NB * D32);

    // ---- Encoder ----
    conv3d_wmma_kernel<32,32><<<cvGrid(D32, 16, NB), B, 0, stream>>>(dataP, Pe1a, t16, 16, 1, noMap);
    conv3d_wmma_kernel<32,32><<<cvGrid(D32, 16, NB), B, 0, stream>>>(t16, Pe1b, x0, 16, 1, noMap);
    maxpool2_kernel<16><<<((NB * D16 * 32) + 255) / 256, B, 0, stream>>>(x0, p0, NB);
    conv3d_wmma_kernel<16,32><<<cvGrid(D16, 32, NB), B, 0, stream>>>(p0, Pe2a, t32, 32, 1, noMap);
    conv3d_wmma_kernel<16,32><<<cvGrid(D16, 20, NB), B, 0, stream>>>(t32, Pe2b, x1, 20, 1, noMap);
    maxpool2_kernel<8><<<((NB * D8 * 32) + 255) / 256, B, 0, stream>>>(x1, xg, NB);

    // ---- Input gates ----
    conv3d_wmma_kernel<8,32><<<cvGrid(D8, 10, NB), B, 0, stream>>>(xg, PW,  w0, 10, 0, noMap);
    conv3d_wmma_kernel<8,32><<<cvGrid(D8, 10, NB), B, 0, stream>>>(xg, PWz, u0, 10, 0, noMap);
    conv3d_wmma_kernel<8,32><<<cvGrid(D8, 10, NB), B, 0, stream>>>(xg, PWr, r0, 10, 0, noMap);

    // ---- Leaf init ----
    leaf_init_kernel<<<((NB * CSp) + 255) / 256, B, 0, stream>>>(w0, u0, h0, height, h, NB);

    // ---- Topological leaf->root propagation ----
    for (int L = 1; L <= maxH; ++L) {
        BatchMap emap; emap.use = 1;
        for (int i = 0; i < 64; ++i) emap.idx[i] = 0;
        Level lv; lv.nNodes = 0;
        int nodeOf[64];
        for (int i = 0; i < 64; ++i) nodeOf[i] = -1;
        for (int nn = 0; nn < 64; ++nn) {
            if (heightH[nn] == L) {
                int li = lv.nNodes++;
                lv.node[li] = nn;
                lv.eA[li] = lv.eB[li] = -1;
                lv.cA[li] = lv.cB[li] = 0;
                nodeOf[nn] = li;
            }
        }
        int E = 0;
        for (int ch = 1; ch < 64; ++ch) {
            int pp = parentH[ch];
            if (heightH[pp] == L) {
                emap.idx[E] = ch;
                int li = nodeOf[pp];
                if (lv.eA[li] < 0) { lv.eA[li] = E; lv.cA[li] = ch; }
                else               { lv.eB[li] = E; lv.cB[li] = ch; }
                ++E;
            }
        }
        conv3d_wmma_kernel<8,32><<<cvGrid(D8, 10, E), B, 0, stream>>>(h, PUr, rjk, 10, 0, emap);
        conv3d_wmma_kernel<8,32><<<cvGrid(D8, 10, E), B, 0, stream>>>(h, PU,  hk,  10, 0, emap);
        level_combine_kernel<<<((lv.nNodes * CSp) + 255) / 256, B, 0, stream>>>(
            rjk, hk, r0, h, lv, hch, wsum);
        conv3d_wmma_kernel<8,32><<<cvGrid(D8, 10, lv.nNodes), B, 0, stream>>>(hch, PUz, upar, 10, 0, noMap);
        level_update_kernel<<<((lv.nNodes * CSp) + 255) / 256, B, 0, stream>>>(
            upar, u0, wsum, w0, hch, lv, h);
    }

    // ---- Decoder ----
    up2_concat_kernel<16,32><<<((NB * D16 * 32) + 255) / 256, B, 0, stream>>>(h, 10, x1, 20, y30, NB);
    conv3d_wmma_kernel<16,32><<<cvGrid(D16, 20, NB), B, 0, stream>>>(y30, Pd2a, y20, 20, 1, noMap);
    conv3d_wmma_kernel<16,32><<<cvGrid(D16, 32, NB), B, 0, stream>>>(y20, Pd2b, y32c, 32, 1, noMap);
    up2_concat_kernel<32,64><<<((NB * D32 * 64) + 255) / 256, B, 0, stream>>>(y32c, 32, x0, 16, y48, NB);
    conv3d_wmma_kernel<32,64><<<cvGrid(D32, 16, NB), B, 0, stream>>>(y48, Pd1a, y16a, 16, 1, noMap);
    conv3d_wmma_kernel<32,32><<<cvGrid(D32, 16, NB), B, 0, stream>>>(y16a, Pd1b, y16b, 16, 1, noMap);
    conv1x1_kernel<<<((NB * D32) + 255) / 256, B, 0, stream>>>(y16b, d1c_w, d1c_b, out, NB * D32);
}
